// lstm_target_40398462386516
// MI455X (gfx1250) — matrix operand order repaired and checked by reading, not compiled, not run on hardware
//
#include <hip/hip_runtime.h>
#include <hip/hip_bf16.h>
#include <stddef.h>

// ---------------------------------------------------------------------------
// Problem constants (from reference)
// ---------------------------------------------------------------------------
#define DIMSZ  1024
#define SLEN   128
#define BATCH  64
#define VOCAB  50257

typedef __attribute__((ext_vector_type(16))) __bf16 v16bf;
typedef __attribute__((ext_vector_type(8)))  __bf16 v8bf;
typedef __attribute__((ext_vector_type(8)))  float  v8f;

// ---------------------------------------------------------------------------
// Helpers
// ---------------------------------------------------------------------------
__device__ __forceinline__ float sigmoidf_(float x) {
    return 1.0f / (1.0f + __expf(-x));
}

__device__ __forceinline__ float tanhf_(float x) {
    x = fminf(15.0f, fmaxf(-15.0f, x));
    float e = __expf(2.0f * x);
    return (e - 1.0f) / (e + 1.0f);
}

__device__ __forceinline__ v8f wmma_bf16(v16bf a, v16bf b, v8f c) {
    // D = A(16x32) * B(32x16) + C, f32 accumulate
    return __builtin_amdgcn_wmma_f32_16x16x32_bf16(
        false, a, false, b, (short)0, c, false, false);
}

// Convert 16 contiguous f32 to a bf16 fragment row (native casts -> packed cvt)
__device__ __forceinline__ v16bf cvt_row16(const float* __restrict__ p) {
    v16bf b;
#pragma unroll
    for (int i = 0; i < 16; ++i) b[i] = (__bf16)p[i];
    return b;
}

// A fragment: 16x32 (MxK) from row-major *bf16* A with leading dim lda.
// Contiguous k order, same as load_b: lanes 0-15 -> M=lane, K = 0..15; lanes 16-31 -> K = 16..31
// Pure 16B vector loads, no conversion.
__device__ __forceinline__ v16bf load_a_bf16(const __bf16* __restrict__ A, int lda,
                                             int mbase, int kbase, int lane) {
    const int row   = mbase + (lane & 15);
    const int khalf = (lane >> 4) * 16;             // 0 or 16
    const __bf16* p = A + (size_t)row * lda + kbase + khalf;
    v8bf lo = *(const v8bf*)p;                      // K = khalf..khalf+7
    v8bf hi = *(const v8bf*)(p + 8);                // K = khalf+8..khalf+15
    return __builtin_shufflevector(lo, hi, 0, 1, 2, 3, 4, 5, 6, 7,
                                           8, 9, 10, 11, 12, 13, 14, 15);
}

// B fragment: 32x16 (KxN) from row-major f32 W[N_total, K] (B = W^T), converted.
// Contiguous k order (as A): col N = lane&15; lanes 0-15 -> K = kbase..kbase+15, lanes 16-31 -> +16
__device__ __forceinline__ v16bf load_b(const float* __restrict__ W, int ldw,
                                        int nrow, int kbase, int lane) {
    const int col = nrow + (lane & 15);
    const int k0  = kbase + ((lane >> 4) << 4);
    return cvt_row16(W + (size_t)col * ldw + k0);
}

// B fragment with K axis split across two weight matrices
// (W1[N,Ks] for k<Ks, W2[N,K-Ks] for k>=Ks). Ks must be a multiple of 16.
__device__ __forceinline__ v16bf load_b_split(const float* __restrict__ W1, int ld1,
                                              const float* __restrict__ W2, int ld2,
                                              int Ks, int nrow, int kbase, int lane) {
    const int col = nrow + (lane & 15);
    const int k0  = kbase + ((lane >> 4) << 4);
    const float* p = (k0 < Ks) ? (W1 + (size_t)col * ld1 + k0)
                               : (W2 + (size_t)col * ld2 + (k0 - Ks));
    return cvt_row16(p);
}

// B fragment with row clamping (ragged VOCAB tail).
__device__ __forceinline__ v16bf load_b_clamp(const float* __restrict__ W, int ldw,
                                              int nrow, int kbase, int lane, int nmax) {
    int col = nrow + (lane & 15);
    col = (col < nmax) ? col : (nmax - 1);
    const int k0 = kbase + ((lane >> 4) << 4);
    return cvt_row16(W + (size_t)col * ldw + k0);
}

// ---------------------------------------------------------------------------
// Kernel 1: masked-softmax attention.  out[b,:] = softmax(context[b]·t[b]) @ context[b]
// one block per batch row, 256 threads (8 waves)
// ---------------------------------------------------------------------------
__global__ void attend_kernel(const float* __restrict__ target,   // [64,1024]
                              const float* __restrict__ context,  // [64,128,1024]
                              float* __restrict__ out) {          // [64,1024]
    const int b   = blockIdx.x;
    const int tid = threadIdx.x;
    __shared__ float t[DIMSZ];
    __shared__ float sc[SLEN];
    __shared__ float red[256];

    for (int k = tid; k < DIMSZ; k += 256) t[k] = target[(size_t)b * DIMSZ + k];
    __syncthreads();

    const int wave = tid >> 5, lane = tid & 31;
    for (int s = wave; s < SLEN; s += 8) {
        const float* cp = context + ((size_t)b * SLEN + s) * DIMSZ;
        float p = 0.0f;
        for (int k = lane; k < DIMSZ; k += 32) p += cp[k] * t[k];
#pragma unroll
        for (int off = 16; off > 0; off >>= 1) p += __shfl_down(p, off, 32);
        if (lane == 0) sc[s] = p;
    }
    __syncthreads();

    // mask (score==0 -> -1e9) + softmax over SLEN
    float x = -3.0e38f;
    if (tid < SLEN) {
        x = sc[tid];
        if (x == 0.0f) x -= 1.0e9f;
    }
    red[tid] = x;
    __syncthreads();
#pragma unroll
    for (int w = 128; w > 0; w >>= 1) {
        if (tid < w) red[tid] = fmaxf(red[tid], red[tid + w]);
        __syncthreads();
    }
    const float m = red[0];
    __syncthreads();
    float e = 0.0f;
    if (tid < SLEN) e = __expf(x - m);
    red[tid] = e;
    __syncthreads();
#pragma unroll
    for (int w = 128; w > 0; w >>= 1) {
        if (tid < w) red[tid] += red[tid + w];
        __syncthreads();
    }
    const float ssum = red[0];
    __syncthreads();
    if (tid < SLEN) sc[tid] = e / ssum;
    __syncthreads();

    // weighted sum over S, coalesced over d; 4 outputs per thread
    float a0 = 0.f, a1 = 0.f, a2 = 0.f, a3 = 0.f;
    const float* cb = context + (size_t)b * SLEN * DIMSZ;
    for (int s = 0; s < SLEN; ++s) {
        const float  p  = sc[s];
        const float* cp = cb + (size_t)s * DIMSZ;
        a0 += p * cp[tid];
        a1 += p * cp[tid + 256];
        a2 += p * cp[tid + 512];
        a3 += p * cp[tid + 768];
    }
    float* ob = out + (size_t)b * DIMSZ;
    ob[tid]       = a0;
    ob[tid + 256] = a1;
    ob[tid + 512] = a2;
    ob[tid + 768] = a3;
}

// ---------------------------------------------------------------------------
// Kernel 2: build bf16 X0 [64,4096] = [embedding | context1 | cls_emb[label] | h0_prev]
// ---------------------------------------------------------------------------
__global__ void build_x0_kernel(const float* __restrict__ emb,
                                const float* __restrict__ ctx1,
                                const float* __restrict__ cls_emb,
                                const int*   __restrict__ label,
                                const float* __restrict__ h0,
                                __bf16* __restrict__ X0) {
    const int idx = blockIdx.x * blockDim.x + threadIdx.x;
    if (idx >= BATCH * 4096) return;
    const int b = idx >> 12, k = idx & 4095;
    float v;
    if (k < 1024)      v = emb[(size_t)b * 1024 + k];
    else if (k < 2048) v = ctx1[(size_t)b * 1024 + (k - 1024)];
    else if (k < 3072) v = cls_emb[(size_t)label[b] * 1024 + (k - 2048)];
    else               v = h0[(size_t)b * 1024 + (k - 3072)];
    X0[idx] = (__bf16)v;
}

// Kernel: copy/convert f32 [64,1024] into a column slice of a bf16 matrix
__global__ void cvt_rows_kernel(const float* __restrict__ src,   // [64,1024]
                                __bf16* __restrict__ dst, int dst_ld, int col_off) {
    const int idx = blockIdx.x * blockDim.x + threadIdx.x;
    if (idx >= BATCH * DIMSZ) return;
    const int b = idx >> 10, k = idx & 1023;
    dst[(size_t)b * dst_ld + col_off + k] = (__bf16)src[idx];
}

// ---------------------------------------------------------------------------
// Kernel 3: LSTM layer — gates GEMM + cell update.
// gates[64,4096] = X[64,K] @ [Wih | Whh]^T ; per wave: one 16-col slice of all 4 gates.
// grid.x = 64 (nt), blockDim = 128 (4 waves = 4 batch tiles).
// Also writes bf16 h into xout[:, xcol..xcol+1023] (next layer's concat input).
// ---------------------------------------------------------------------------
__global__ void lstm_layer_kernel(const __bf16* __restrict__ X,   // [64, K] bf16
                                  int K, int Ksplit,
                                  const float* __restrict__ Wih,  // [4096, Ksplit]
                                  const float* __restrict__ Whh,  // [4096, K-Ksplit]
                                  const float* __restrict__ c_prev, // [64,1024]
                                  float* __restrict__ h_out,      // [64,1024]
                                  float* __restrict__ c_out,      // [64,1024]
                                  __bf16* __restrict__ xout, int xout_ld, int xcol) {
    const int lane = threadIdx.x & 31;
    const int mt   = threadIdx.x >> 5;     // 0..3  batch tile
    const int nt   = blockIdx.x;           // 0..63 gate-dim tile
    const int Khh  = K - Ksplit;

    v8f acc_i = {}, acc_f = {}, acc_g = {}, acc_o = {};
    for (int kb = 0; kb < K; kb += 32) {
        v16bf a  = load_a_bf16(X, K, mt * 16, kb, lane);
        v16bf bi = load_b_split(Wih, Ksplit, Whh, Khh, Ksplit, 0    + nt * 16, kb, lane);
        v16bf bf = load_b_split(Wih, Ksplit, Whh, Khh, Ksplit, 1024 + nt * 16, kb, lane);
        v16bf bg = load_b_split(Wih, Ksplit, Whh, Khh, Ksplit, 2048 + nt * 16, kb, lane);
        v16bf bo = load_b_split(Wih, Ksplit, Whh, Khh, Ksplit, 3072 + nt * 16, kb, lane);
        acc_i = wmma_bf16(a, bi, acc_i);
        acc_f = wmma_bf16(a, bf, acc_f);
        acc_g = wmma_bf16(a, bg, acc_g);
        acc_o = wmma_bf16(a, bo, acc_o);
    }

    const int col   = nt * 16 + (lane & 15);
    const int rbase = mt * 16 + ((lane >> 4) << 3);
#pragma unroll
    for (int r = 0; r < 8; ++r) {
        const int brow = rbase + r;
        const float iv = sigmoidf_(acc_i[r]);
        const float fv = sigmoidf_(acc_f[r]);
        const float gv = tanhf_(acc_g[r]);
        const float ov = sigmoidf_(acc_o[r]);
        const float cn = fv * c_prev[(size_t)brow * DIMSZ + col] + iv * gv;
        const float hn = ov * tanhf_(cn);
        c_out[(size_t)brow * DIMSZ + col] = cn;
        h_out[(size_t)brow * DIMSZ + col] = hn;
        xout[(size_t)brow * xout_ld + xcol + col] = (__bf16)hn;
    }
}

// ---------------------------------------------------------------------------
// Kernel 4: attv = tanh( X[64,2048] @ W_att^T )  (W_att [1024,2048]) -> bf16
// grid.x = 64 (nt), blockDim = 128 (4 waves = batch tiles)
// ---------------------------------------------------------------------------
__global__ void att_gemm_kernel(const __bf16* __restrict__ X,
                                const float* __restrict__ W,
                                __bf16* __restrict__ out) {   // [64,1024] bf16
    const int lane = threadIdx.x & 31;
    const int mt   = threadIdx.x >> 5;
    const int nt   = blockIdx.x;
    v8f acc = {};
    for (int kb = 0; kb < 2048; kb += 32) {
        v16bf a = load_a_bf16(X, 2048, mt * 16, kb, lane);
        v16bf b = load_b(W, 2048, nt * 16, kb, lane);
        acc = wmma_bf16(a, b, acc);
    }
    const int col   = nt * 16 + (lane & 15);
    const int rbase = mt * 16 + ((lane >> 4) << 3);
#pragma unroll
    for (int r = 0; r < 8; ++r)
        out[(size_t)(rbase + r) * DIMSZ + col] = (__bf16)tanhf_(acc[r]);
}

// ---------------------------------------------------------------------------
// Kernel 5: pred[64,50257] = A[64,1024](bf16) @ W_out^T.  One wave per 16-wide
// vocab slice; B fragment reused across the 4 batch tiles (B traffic dominates).
// ---------------------------------------------------------------------------
__global__ void pred_gemm_kernel(const __bf16* __restrict__ A,  // [64,1024] bf16
                                 const float* __restrict__ W,   // [50257,1024]
                                 float* __restrict__ out) {     // [64,50257]
    const int lane = threadIdx.x & 31;
    const int w    = threadIdx.x >> 5;
    const int nt   = blockIdx.x * 4 + w;
    const int NT   = (VOCAB + 15) / 16;   // 3142
    if (nt >= NT) return;                 // wave-uniform exit -> EXEC stays full

    // prefetch pointer for this lane's weight row (clamped)
    int pcol = nt * 16 + (lane & 15);
    pcol = (pcol < VOCAB) ? pcol : (VOCAB - 1);
    const float* wrow = W + (size_t)pcol * DIMSZ;

    v8f acc0 = {}, acc1 = {}, acc2 = {}, acc3 = {};
    for (int kb = 0; kb < DIMSZ; kb += 32) {
        if (kb + 64 < DIMSZ) __builtin_prefetch(wrow + kb + 64, 0, 0);
        v16bf bfr = load_b_clamp(W, DIMSZ, nt * 16, kb, lane, VOCAB);
        v16bf a0  = load_a_bf16(A, DIMSZ, 0,  kb, lane);
        v16bf a1  = load_a_bf16(A, DIMSZ, 16, kb, lane);
        v16bf a2  = load_a_bf16(A, DIMSZ, 32, kb, lane);
        v16bf a3  = load_a_bf16(A, DIMSZ, 48, kb, lane);
        acc0 = wmma_bf16(a0, bfr, acc0);
        acc1 = wmma_bf16(a1, bfr, acc1);
        acc2 = wmma_bf16(a2, bfr, acc2);
        acc3 = wmma_bf16(a3, bfr, acc3);
    }

    const int col = nt * 16 + (lane & 15);
    if (col < VOCAB) {
        const int rb = (lane >> 4) << 3;
#pragma unroll
        for (int r = 0; r < 8; ++r) {
            out[(size_t)(0  + rb + r) * VOCAB + col] = acc0[r];
            out[(size_t)(16 + rb + r) * VOCAB + col] = acc1[r];
            out[(size_t)(32 + rb + r) * VOCAB + col] = acc2[r];
            out[(size_t)(48 + rb + r) * VOCAB + col] = acc3[r];
        }
    }
}

// ---------------------------------------------------------------------------
// Host launcher
// ---------------------------------------------------------------------------
extern "C" void kernel_launch(void* const* d_in, const int* in_sizes, int n_in,
                              void* d_out, int out_size, void* d_ws, size_t ws_size,
                              hipStream_t stream) {
    (void)in_sizes; (void)n_in; (void)out_size; (void)ws_size;

    const float* context  = (const float*)d_in[0];   // [64,128,1024]
    const float* h_prev   = (const float*)d_in[1];   // [2,64,1024]
    const float* c_prev   = (const float*)d_in[2];   // [2,64,1024]
    const float* embedding= (const float*)d_in[3];   // [64,1024]
    const int*   cls_label= (const int*)  d_in[4];   // [64]
    const float* W_ih0    = (const float*)d_in[5];   // [4096,3072]
    const float* W_hh0    = (const float*)d_in[6];   // [4096,1024]
    const float* W_ih1    = (const float*)d_in[7];   // [4096,1024]
    const float* W_hh1    = (const float*)d_in[8];   // [4096,1024]
    const float* cls_emb  = (const float*)d_in[9];   // [4,1024]
    const float* W_att    = (const float*)d_in[10];  // [1024,2048]
    const float* W_out    = (const float*)d_in[11];  // [50257,1024]

    // output layout: pred | h_new | c_new
    float* out    = (float*)d_out;
    float* pred   = out;                                  // 64*50257
    float* h_out  = out + (size_t)BATCH * VOCAB;          // 2*64*1024
    float* c_out  = h_out + 2 * BATCH * DIMSZ;
    float* h0_out = h_out;
    float* h1_out = h_out + BATCH * DIMSZ;
    float* c0_out = c_out;
    float* c1_out = c_out + BATCH * DIMSZ;

    // workspace layout
    char* ws = (char*)d_ws;
    float*  ctx1 = (float*)  ws;                      ws += BATCH * DIMSZ * 4;  // f32
    float*  comb = (float*)  ws;                      ws += BATCH * DIMSZ * 4;  // f32
    __bf16* X0   = (__bf16*) ws;                      ws += BATCH * 4096 * 2;   // bf16
    __bf16* X1   = (__bf16*) ws;                      ws += BATCH * 2048 * 2;   // bf16
    __bf16* Xatt = (__bf16*) ws;                      ws += BATCH * 2048 * 2;   // bf16
    __bf16* attv = (__bf16*) ws;                      ws += BATCH * DIMSZ * 2;  // bf16

    const float* h0_prev = h_prev;
    const float* h1_prev = h_prev + BATCH * DIMSZ;
    const float* c0_prev = c_prev;
    const float* c1_prev = c_prev + BATCH * DIMSZ;

    // 1. context1 = masked_attend(h[-1], context)
    attend_kernel<<<BATCH, 256, 0, stream>>>(h1_prev, context, ctx1);

    // 2. X1[:,1024:2048] = bf16(h1_prev)   (independent of step 1)
    cvt_rows_kernel<<<(BATCH * DIMSZ) / 256, 256, 0, stream>>>(h1_prev, X1, 2048, 1024);

    // 3. X0 = bf16([embedding | context1 | cls_emb[label] | h0_prev])
    build_x0_kernel<<<(BATCH * 4096) / 256, 256, 0, stream>>>(
        embedding, ctx1, cls_emb, cls_label, h0_prev, X0);

    // 4. layer 0 LSTM cell (K=4096, split at 3072); bf16 h -> X1[:,0:1024]
    lstm_layer_kernel<<<64, 128, 0, stream>>>(X0, 4096, 3072, W_ih0, W_hh0,
                                              c0_prev, h0_out, c0_out,
                                              X1, 2048, 0);

    // 5. layer 1 LSTM cell (K=2048, split at 1024); bf16 h -> Xatt[:,1024:2048]
    lstm_layer_kernel<<<64, 128, 0, stream>>>(X1, 2048, 1024, W_ih1, W_hh1,
                                              c1_prev, h1_out, c1_out,
                                              Xatt, 2048, 1024);

    // 6. combined = masked_attend(h1_new, context)
    attend_kernel<<<BATCH, 256, 0, stream>>>(h1_out, context, comb);

    // 7. Xatt[:,0:1024] = bf16(combined)
    cvt_rows_kernel<<<(BATCH * DIMSZ) / 256, 256, 0, stream>>>(comb, Xatt, 2048, 0);

    // 8. attv = bf16(tanh(Xatt @ W_att^T))
    att_gemm_kernel<<<64, 128, 0, stream>>>(Xatt, W_att, attv);

    // 9. pred = attv @ W_out^T
    const int NT = (VOCAB + 15) / 16;                 // 3142
    pred_gemm_kernel<<<(NT + 3) / 4, 128, 0, stream>>>(attv, W_out, pred);
}
